// GPTBlock_4294967296500
// MI455X (gfx1250) — compile-verified, matrix operand order checked by reading, not run on hardware
//
#include <hip/hip_runtime.h>
#include <hip/hip_bf16.h>

// ---------------------------------------------------------------- types
typedef _Float16 h16;
typedef __attribute__((ext_vector_type(16))) _Float16 v16h;
typedef __attribute__((ext_vector_type(8)))  _Float16 v8h;
typedef __attribute__((ext_vector_type(8)))  float    v8f;
typedef __attribute__((ext_vector_type(4)))  unsigned int u32x4;
typedef __attribute__((ext_vector_type(8)))  unsigned int u32x8;

// Model constants
#define NBATCH 2
#define SEQ    2048
#define NTOK   4096      // NBATCH*SEQ
#define DM     1024
#define NH     16
#define DK     64
#define DFF    2752

__device__ __forceinline__ v16h ld_frag(const h16* p) {
    // CDNA5 16-bit A/B fragment: halves j<8 at [hi*8 + j], j>=8 at [16 + hi*8 + j-8]
    v8h a = *(const v8h*)(p);
    v8h b = *(const v8h*)(p + 16);
    return __builtin_shufflevector(a, b, 0,1,2,3,4,5,6,7,8,9,10,11,12,13,14,15);
}

__device__ __forceinline__ v8f wmma_f16(v16h a, v16h b, v8f c) {
    return __builtin_amdgcn_wmma_f32_16x16x32_f16(
        /*neg_a=*/false, a, /*neg_b=*/false, b,
        /*c_mod=*/(short)0, c, /*reuse_a=*/false, /*reuse_b=*/false);
}

// ---------------------------------------------------------------- TDM 2D tile load
// Issues TENSOR_LOAD_TO_LDS (wave-level DMA, tracked by TENSORcnt).
// D# group0: count=1 | lds_addr | global_addr | type=2 ("image").
// D# group1: data_size=2B, tensor_dim0, tensor_dim1(large), tile_dim0/1,
//            tensor_dim0_stride. 2D form: descriptor groups 2/3 omitted.
__device__ __forceinline__ void tdm_load_2d(unsigned lds_off, const void* gaddr,
                                            unsigned tile_d0, unsigned tile_d1,
                                            unsigned tensor_d0, unsigned stride0) {
    unsigned long long ga = (unsigned long long)(uintptr_t)gaddr;
    u32x4 g0;
    g0[0] = 1u;                                       // count=1, user descriptor
    g0[1] = lds_off;                                  // lds_addr (bytes)
    g0[2] = (unsigned)(ga & 0xFFFFFFFFu);             // global_addr[31:0]
    g0[3] = (unsigned)((ga >> 32) & 0x1FFFFFFu)       // global_addr[56:32]
            | (2u << 30);                             // type = 2
    u32x8 g1;
    g1[0] = (1u << 16);                               // data_size=1 (2 bytes)
    g1[1] = (tensor_d0 & 0xFFFFu) << 16;              // tensor_dim0[15:0]
    g1[2] = ((tensor_d0 >> 16) & 0xFFFFu)             // tensor_dim0[31:16]
            | (0xFFFFu << 16);                        // tensor_dim1[15:0]
    g1[3] = 0x7FFFu                                   // tensor_dim1[31:16] (large)
            | ((tile_d0 & 0xFFFFu) << 16);            // tile_dim0
    g1[4] = (tile_d1 & 0xFFFFu);                      // tile_dim1 (tile_dim2=0)
    g1[5] = stride0;                                  // tensor_dim0_stride[31:0]
    g1[6] = 0;                                        // stride0 hi | stride1 lo
    g1[7] = 0;                                        // stride1 hi
    asm volatile("tensor_load_to_lds %0, %1" :: "s"(g0), "s"(g1) : "memory");
}

// ---------------------------------------------------------------- f32 -> f16 convert
__global__ __launch_bounds__(256) void k_cvt(const float* __restrict__ src,
                                             h16* __restrict__ dst, int n) {
    int i = (blockIdx.x * 256 + threadIdx.x) * 4;
    if (i + 3 < n) {
        float4 v = *(const float4*)(src + i);
        dst[i + 0] = (h16)v.x; dst[i + 1] = (h16)v.y;
        dst[i + 2] = (h16)v.z; dst[i + 3] = (h16)v.w;
    }
}

// ---------------------------------------------------------------- RMSNorm (row of 1024) -> f16
__global__ __launch_bounds__(256) void k_rmsnorm(const float* __restrict__ x,
                                                 const float* __restrict__ wt,
                                                 h16* __restrict__ out) {
    __shared__ float red[8];
    const size_t row = blockIdx.x;
    const int tid = threadIdx.x, lane = tid & 31, w = tid >> 5;
    const float* xr = x + row * DM;
    float4 v = *(const float4*)(xr + tid * 4);
    float ss = v.x*v.x + v.y*v.y + v.z*v.z + v.w*v.w;
#pragma unroll
    for (int off = 1; off < 32; off <<= 1) ss += __shfl_xor(ss, off, 32);
    if (lane == 0) red[w] = ss;
    __syncthreads();
    float tot = 0.f;
#pragma unroll
    for (int i = 0; i < 8; ++i) tot += red[i];
    float rinv = 1.0f / sqrtf(tot * (1.0f / (float)DM) + 1e-10f);
    const float* wp = wt + tid * 4;
    h16* op = out + row * DM + tid * 4;
    op[0] = (h16)(wp[0] * v.x * rinv);
    op[1] = (h16)(wp[1] * v.y * rinv);
    op[2] = (h16)(wp[2] * v.z * rinv);
    op[3] = (h16)(wp[3] * v.w * rinv);
}

// ---------------------------------------------------------------- generic WMMA GEMM
// C[M,N] = A[M,K] * B[N,K]^T ; A,B f16 row-major K-contiguous.
// Tiles staged into LDS by the Tensor Data Mover (wave 0 issues, TENSORcnt drains).
// EPI 0: store f16 into Cout ; EPI 2: store f32 Cout = acc + resid
template<int EPI>
__global__ __launch_bounds__(256) void k_gemm(const h16* __restrict__ A,
                                              const h16* __restrict__ Bm,
                                              void* __restrict__ Cout,
                                              const float* __restrict__ resid,
                                              int M, int N, int K) {
    __shared__ __align__(128) h16 As[256 * 32];
    __shared__ __align__(128) h16 Bs[64 * 32];
    const int tid = threadIdx.x;
    const int w = tid >> 5, lane = tid & 31, lo = lane & 15, hi = lane >> 4;
    const int mBlk = blockIdx.y * 256;
    const int nBlk = blockIdx.x * 64;
    const unsigned ldsA = (unsigned)(uintptr_t)(void*)As;  // LDS byte offset
    const unsigned ldsB = (unsigned)(uintptr_t)(void*)Bs;

    v8f acc[2][4] = {};

    for (int k0 = 0; k0 < K; k0 += 32) {
        if (w == 0) {  // TDM: DMA both tiles into LDS, then drain TENSORcnt
            tdm_load_2d(ldsA, A + (size_t)mBlk * K + k0, 32, 256, (unsigned)K,
                        (unsigned)K);
            tdm_load_2d(ldsB, Bm + (size_t)nBlk * K + k0, 32, 64, (unsigned)K,
                        (unsigned)K);
            __builtin_amdgcn_s_wait_tensorcnt(0);
        }
        if (k0 + 32 < K)  // warm next A rows into L2 (global_prefetch_b8)
            __builtin_prefetch(A + (size_t)(mBlk + (tid >> 2)) * K + k0 + 32, 0, 1);
        __syncthreads();

        v16h bfrag[4];
#pragma unroll
        for (int nt = 0; nt < 4; ++nt)
            bfrag[nt] = ld_frag(&Bs[(nt * 16 + lo) * 32 + hi * 8]);
#pragma unroll
        for (int mt = 0; mt < 2; ++mt) {
            v16h afrag = ld_frag(&As[(w * 32 + mt * 16 + lo) * 32 + hi * 8]);
#pragma unroll
            for (int nt = 0; nt < 4; ++nt)
                acc[mt][nt] = wmma_f16(afrag, bfrag[nt], acc[mt][nt]);
        }
        __syncthreads();
    }

    // epilogue (C layout: VGPR r -> row = r + 8*hi, col = lo within 16x16 tile)
#pragma unroll
    for (int mt = 0; mt < 2; ++mt)
#pragma unroll
        for (int nt = 0; nt < 4; ++nt)
#pragma unroll
            for (int r = 0; r < 8; ++r) {
                size_t row = (size_t)(mBlk + w * 32 + mt * 16 + hi * 8 + r);
                size_t col = (size_t)(nBlk + nt * 16 + lo);
                float vv = acc[mt][nt][r];
                if (EPI == 0) {
                    ((h16*)Cout)[row * N + col] = (h16)vv;
                } else {
                    ((float*)Cout)[row * N + col] = vv + resid[row * N + col];
                }
            }
}

// ---------------------------------------------------------------- QK rmsnorm + RoPE + V transpose
// Wave handles one (b,h,s). lane owns dims {lane, lane+32} (the RoPE rotate-half pair).
__global__ __launch_bounds__(256) void k_qkprep(const h16* __restrict__ q,
                                                const h16* __restrict__ k,
                                                const h16* __restrict__ v,
                                                const float* __restrict__ qw,
                                                const float* __restrict__ kw,
                                                h16* __restrict__ qh,
                                                h16* __restrict__ kh,
                                                h16* __restrict__ vt) {
    const int w = threadIdx.x >> 5, lane = threadIdx.x & 31;
    const int item = blockIdx.x * 8 + w;           // [0, NBATCH*NH*SEQ)
    const int b = item >> 15;                      // /(NH*SEQ)
    const int h = (item >> 11) & 15;
    const int s = item & 2047;
    const size_t token = (size_t)b * SEQ + s;
    const size_t bh = (size_t)b * NH + h;
    const int d0 = lane, d1 = lane + 32;
    const size_t base = token * DM + h * DK;

    float ifr = __expf(-(float)lane * (1.0f / 32.0f) * logf(500000.0f));
    float ang = (float)s * ifr;
    float cs = __cosf(ang), sn = __sinf(ang);

#pragma unroll
    for (int which = 0; which < 2; ++which) {
        const h16* src = which ? k : q;
        const float* nw = which ? kw : qw;
        h16* dst = which ? kh : qh;
        float x0 = (float)src[base + d0];
        float x1 = (float)src[base + d1];
        float ss = x0 * x0 + x1 * x1;
#pragma unroll
        for (int off = 1; off < 32; off <<= 1) ss += __shfl_xor(ss, off, 32);
        float rinv = 1.0f / sqrtf(ss * (1.0f / (float)DK) + 1e-10f);
        float n0 = nw[d0] * x0 * rinv;
        float n1 = nw[d1] * x1 * rinv;
        float r0 = n0 * cs - n1 * sn;   // x*cos + rotate_half(x)*sin
        float r1 = n1 * cs + n0 * sn;
        h16* dp = dst + (bh * SEQ + s) * DK;
        dp[d0] = (h16)r0;
        dp[d1] = (h16)r1;
    }
    // V transposed to [b,h,d,s] for B-fragments of P*V
    vt[(bh * DK + d0) * SEQ + s] = v[base + d0];
    vt[(bh * DK + d1) * SEQ + s] = v[base + d1];
}

// ---------------------------------------------------------------- flash attention (causal)
// Block = (b, h, 64-query tile); 4 waves x 16 queries. WMMA scores + WMMA P*V.
__global__ __launch_bounds__(128) void k_attn(const h16* __restrict__ qh,
                                              const h16* __restrict__ kh,
                                              const h16* __restrict__ vt,
                                              h16* __restrict__ ao) {
    __shared__ h16 Pl[4][16][64];
    const int w = threadIdx.x >> 5, lane = threadIdx.x & 31;
    const int lo = lane & 15, hi = lane >> 4;
    const int b = blockIdx.z, h = blockIdx.y, qt = blockIdx.x;
    const size_t bh = (size_t)b * NH + h;
    const int qbase = qt * 64 + w * 16;

    // Q fragments (A operand), 2 chunks of K-dim 32
    v16h qf[2];
#pragma unroll
    for (int dc = 0; dc < 2; ++dc)
        qf[dc] = ld_frag(qh + (bh * SEQ + qbase + lo) * DK + dc * 32 + hi * 8);

    v8f oacc[4] = {};
    float mrow[8], lrow[8];
#pragma unroll
    for (int r = 0; r < 8; ++r) { mrow[r] = -3.0e38f; lrow[r] = 0.0f; }

    for (int kt = 0; kt <= qt; ++kt) {
        v8f sacc[4] = {};
#pragma unroll
        for (int dc = 0; dc < 2; ++dc)
#pragma unroll
            for (int nt = 0; nt < 4; ++nt) {
                v16h kf = ld_frag(kh + (bh * SEQ + kt * 64 + nt * 16 + lo) * DK +
                                  dc * 32 + hi * 8);
                sacc[nt] = wmma_f16(qf[dc], kf, sacc[nt]);
            }
        const bool diag = (kt == qt);
#pragma unroll
        for (int r = 0; r < 8; ++r) {
            const int qrow = qbase + hi * 8 + r;
            float mx = -3.0e38f;
#pragma unroll
            for (int nt = 0; nt < 4; ++nt) {
                float sv = sacc[nt][r] * 0.125f;           // 1/sqrt(64)
                int key = kt * 64 + nt * 16 + lo;
                if (diag && key > qrow) sv = -3.0e38f;     // causal mask
                sacc[nt][r] = sv;
                mx = fmaxf(mx, sv);
            }
#pragma unroll
            for (int off = 1; off < 16; off <<= 1)         // reduce over 16 cols
                mx = fmaxf(mx, __shfl_xor(mx, off, 32));
            float mnew = fmaxf(mrow[r], mx);
            float corr = __expf(mrow[r] - mnew);
            float psum = 0.0f;
#pragma unroll
            for (int nt = 0; nt < 4; ++nt) {
                float p = __expf(sacc[nt][r] - mnew);
                psum += p;
                Pl[w][hi * 8 + r][nt * 16 + lo] = (h16)p;
            }
#pragma unroll
            for (int off = 1; off < 16; off <<= 1)
                psum += __shfl_xor(psum, off, 32);
            lrow[r] = lrow[r] * corr + psum;
            mrow[r] = mnew;
#pragma unroll
            for (int nt = 0; nt < 4; ++nt) oacc[nt][r] *= corr;
        }
        // wave-synchronous LDS: drain DS pipe before cross-lane fragment reload
        asm volatile("s_wait_dscnt 0x0" ::: "memory");
#pragma unroll
        for (int kc = 0; kc < 2; ++kc) {
            v16h pf = ld_frag(&Pl[w][lo][kc * 32 + hi * 8]);
#pragma unroll
            for (int nt = 0; nt < 4; ++nt) {
                v16h vf = ld_frag(vt + (bh * DK + nt * 16 + lo) * SEQ +
                                  kt * 64 + kc * 32 + hi * 8);
                oacc[nt] = wmma_f16(pf, vf, oacc[nt]);
            }
        }
    }
    // write attention output in [b, s, h*64 + d] layout (f16 for Wo GEMM)
#pragma unroll
    for (int r = 0; r < 8; ++r) {
        float rinv = 1.0f / lrow[r];
        size_t srow = (size_t)b * SEQ + qbase + hi * 8 + r;
#pragma unroll
        for (int nt = 0; nt < 4; ++nt)
            ao[srow * DM + h * DK + nt * 16 + lo] = (h16)(oacc[nt][r] * rinv);
    }
}

// ---------------------------------------------------------------- SwiGLU gate
__global__ __launch_bounds__(256) void k_gate(const h16* __restrict__ a,
                                              const h16* __restrict__ bb,
                                              h16* __restrict__ g, int n) {
    int i = (blockIdx.x * 256 + threadIdx.x) * 4;
    if (i + 3 < n) {
#pragma unroll
        for (int j = 0; j < 4; ++j) {
            float av = (float)a[i + j];
            float bv = (float)bb[i + j];
            float s = av / (1.0f + __expf(-av));   // silu
            g[i + j] = (h16)(s * bv);
        }
    }
}

// ---------------------------------------------------------------- launcher
extern "C" void kernel_launch(void* const* d_in, const int* in_sizes, int n_in,
                              void* d_out, int out_size, void* d_ws, size_t ws_size,
                              hipStream_t stream) {
    (void)in_sizes; (void)n_in; (void)out_size; (void)ws_size;
    const float* x     = (const float*)d_in[0];
    const float* wq    = (const float*)d_in[1];
    const float* wk    = (const float*)d_in[2];
    const float* wv    = (const float*)d_in[3];
    const float* wo    = (const float*)d_in[4];
    const float* qnw   = (const float*)d_in[5];
    const float* knw   = (const float*)d_in[6];
    const float* n1w   = (const float*)d_in[7];
    const float* n2w   = (const float*)d_in[8];
    const float* w1    = (const float*)d_in[9];
    const float* w3    = (const float*)d_in[10];
    const float* w2    = (const float*)d_in[11];
    float* out = (float*)d_out;

    char* ws = (char*)d_ws;
    size_t off = 0;
    auto alloc = [&](size_t bytes) {
        size_t o = off;
        off = (off + bytes + 255) & ~(size_t)255;
        return o;
    };
    h16* xn1  = (h16*)(ws + alloc((size_t)NTOK * DM * 2));
    h16* xn2  = (h16*)(ws + alloc((size_t)NTOK * DM * 2));
    h16* wqf  = (h16*)(ws + alloc((size_t)DM * DM * 2));
    h16* wkf  = (h16*)(ws + alloc((size_t)DM * DM * 2));
    h16* wvf  = (h16*)(ws + alloc((size_t)DM * DM * 2));
    h16* wof  = (h16*)(ws + alloc((size_t)DM * DM * 2));
    h16* w1f  = (h16*)(ws + alloc((size_t)DFF * DM * 2));
    h16* w3f  = (h16*)(ws + alloc((size_t)DFF * DM * 2));
    h16* w2f  = (h16*)(ws + alloc((size_t)DM * DFF * 2));
    h16* qf   = (h16*)(ws + alloc((size_t)NTOK * DM * 2));
    h16* kf   = (h16*)(ws + alloc((size_t)NTOK * DM * 2));
    h16* vf   = (h16*)(ws + alloc((size_t)NTOK * DM * 2));
    h16* qhd  = (h16*)(ws + alloc((size_t)NTOK * DM * 2));
    h16* khd  = (h16*)(ws + alloc((size_t)NTOK * DM * 2));
    h16* vtd  = (h16*)(ws + alloc((size_t)NTOK * DM * 2));
    h16* ao   = (h16*)(ws + alloc((size_t)NTOK * DM * 2));
    float* hb = (float*)(ws + alloc((size_t)NTOK * DM * 4));
    h16* a16  = (h16*)(ws + alloc((size_t)NTOK * DFF * 2));
    h16* b16  = (h16*)(ws + alloc((size_t)NTOK * DFF * 2));
    h16* g16  = (h16*)(ws + alloc((size_t)NTOK * DFF * 2));

    // weight conversions
    const int nDM2 = DM * DM, nFF = DFF * DM;
    k_cvt<<<nDM2 / 1024, 256, 0, stream>>>(wq, wqf, nDM2);
    k_cvt<<<nDM2 / 1024, 256, 0, stream>>>(wk, wkf, nDM2);
    k_cvt<<<nDM2 / 1024, 256, 0, stream>>>(wv, wvf, nDM2);
    k_cvt<<<nDM2 / 1024, 256, 0, stream>>>(wo, wof, nDM2);
    k_cvt<<<nFF  / 1024, 256, 0, stream>>>(w1, w1f, nFF);
    k_cvt<<<nFF  / 1024, 256, 0, stream>>>(w3, w3f, nFF);
    k_cvt<<<nFF  / 1024, 256, 0, stream>>>(w2, w2f, nFF);

    // attention path
    k_rmsnorm<<<NTOK, 256, 0, stream>>>(x, n1w, xn1);
    dim3 gQKV(DM / 64, NTOK / 256);
    k_gemm<0><<<gQKV, 256, 0, stream>>>(xn1, wqf, qf, nullptr, NTOK, DM, DM);
    k_gemm<0><<<gQKV, 256, 0, stream>>>(xn1, wkf, kf, nullptr, NTOK, DM, DM);
    k_gemm<0><<<gQKV, 256, 0, stream>>>(xn1, wvf, vf, nullptr, NTOK, DM, DM);
    k_qkprep<<<(NBATCH * NH * SEQ) / 8, 256, 0, stream>>>(qf, kf, vf, qnw, knw,
                                                          qhd, khd, vtd);
    dim3 gAtt(SEQ / 64, NH, NBATCH);
    k_attn<<<gAtt, 128, 0, stream>>>(qhd, khd, vtd, ao);
    k_gemm<2><<<gQKV, 256, 0, stream>>>(ao, wof, hb, x, NTOK, DM, DM);  // h = x + attn

    // FFN path
    k_rmsnorm<<<NTOK, 256, 0, stream>>>(hb, n2w, xn2);
    dim3 gFF(DFF / 64, NTOK / 256);
    k_gemm<0><<<gFF, 256, 0, stream>>>(xn2, w1f, a16, nullptr, NTOK, DFF, DM);
    k_gemm<0><<<gFF, 256, 0, stream>>>(xn2, w3f, b16, nullptr, NTOK, DFF, DM);
    const int nGate = NTOK * DFF;
    k_gate<<<nGate / 1024, 256, 0, stream>>>(a16, b16, g16, nGate);
    dim3 gOut(DM / 64, NTOK / 256);
    k_gemm<2><<<gOut, 256, 0, stream>>>(g16, w2f, out, hb, NTOK, DM, DFF); // y = h + ffn
}
